// ImprovedGNN_20186346291721
// MI455X (gfx1250) — matrix operand order repaired and checked by reading, not compiled, not run on hardware
//
#include <hip/hip_runtime.h>

// ---------------------------------------------------------------------------
// ImprovedGNN on MI455X (gfx1250): 2-layer GraphSAGE, H = 128 everywhere.
//   - genre masked-mean pool + concat -> h (f16)     (mem-bound, trivial)
//   - per-layer: edge scatter, f16 gather + f32 L2 atomics   (dominant)
//   - per-layer: fused dual-GEMM + bias+BN+ReLU (v_wmma_f32_16x16x32_f16)
//   - head: dot(128) + sigmoid + rescale
//
// Activations are kept in f16 end-to-end (they feed f16 WMMA anyway), halving
// the dominant scatter-gather traffic; the neighbor-sum stays in f32 atomics.
// WMMA fragments are fully vectorized:
//   B: row-major f16 W[h][k] -> one aligned v16h (2x global_load_b128) per frag
//   A: LDS-staged f16 rows  -> 2x ds_load_b128 per frag
// ---------------------------------------------------------------------------

typedef __attribute__((ext_vector_type(16))) _Float16 v16h;
typedef __attribute__((ext_vector_type(8)))  _Float16 v8h;
typedef __attribute__((ext_vector_type(8)))  float    v8f;

#define HC 128   // hidden/feature width (D0 == H == 128)
#define MG 10
#define DE 32
#define ING 96

// ---------------------------------------------------------------------------
// Zero fill
__global__ void zero_kernel(float* __restrict__ p, long long n) {
  long long i = (long long)blockIdx.x * blockDim.x + threadIdx.x;
  if (i < n) p[i] = 0.0f;
}

// ---------------------------------------------------------------------------
// Weight prep: f32 -> f16, SAME row-major [h][k] layout. A lane's B fragment
// (16 consecutive K of one output channel) is then 32 contiguous bytes.
__global__ void wprep_kernel(const float* __restrict__ W, _Float16* __restrict__ Wh) {
  int idx = blockIdx.x * blockDim.x + threadIdx.x;
  if (idx >= HC * HC / 4) return;
  const float4 w = ((const float4*)W)[idx];
  _Float16* o = (_Float16*)Wh + idx * 4;
  o[0] = (_Float16)w.x; o[1] = (_Float16)w.y;
  o[2] = (_Float16)w.z; o[3] = (_Float16)w.w;
}

// ---------------------------------------------------------------------------
// Genre masked-mean pool + concat -> h0[N][128] (f16)
__global__ void pool_concat_kernel(const float* __restrict__ x,
                                   const float* __restrict__ gemb,
                                   const int* __restrict__ gids,
                                   const unsigned char* __restrict__ gmask,
                                   _Float16* __restrict__ h0, int N) {
  int idx = blockIdx.x * blockDim.x + threadIdx.x;
  if (idx >= N * HC) return;
  int n = idx >> 7;
  int f = idx & 127;
  float v;
  if (f < ING) {
    v = x[n * ING + f];
  } else {
    int d = f - ING;
    float acc = 0.0f, c = 0.0f;
#pragma unroll
    for (int j = 0; j < MG; ++j) {
      if (gmask[n * MG + j]) {
        acc += gemb[gids[n * MG + j] * DE + d];
        c += 1.0f;
      }
    }
    v = (c > 0.0f) ? acc / (c + 1e-8f) : 0.0f;
  }
  h0[idx] = (_Float16)v;
}

// ---------------------------------------------------------------------------
// Edge scatter: agg[dst] += h[src] (f32 accumulate), cnt[dst] += 1.
// One thread per (edge, 8-f16 chunk): one 16B gather + 8 HW f32 atomics.
// Targets are L2-resident (12.8/25.6 MB << 192 MB L2), non-returning atomics.
__global__ void edge_scatter_kernel(const _Float16* __restrict__ h,
                                    const int* __restrict__ ei,
                                    float* __restrict__ agg,
                                    float* __restrict__ cnt, int E) {
  long long idx = (long long)blockIdx.x * blockDim.x + threadIdx.x;
  if (idx >= (long long)E * 16) return;
  int e  = (int)(idx >> 4);
  int fc = (int)(idx & 15);
  int s = ei[e];
  int d = ei[E + e];
  const v8h v = *(const v8h*)(h + (long long)s * HC + fc * 8);
  float* a = agg + (long long)d * HC + fc * 8;
#pragma unroll
  for (int j = 0; j < 8; ++j) unsafeAtomicAdd(a + j, (float)v[j]);
  if (fc == 0) unsafeAtomicAdd(cnt + d, 1.0f);
}

// ---------------------------------------------------------------------------
// Fused SAGE layer: out = ReLU(BN(mean@WlT + b + h@WrT)), out in f16.
// Block = 256 threads = 8 waves; block owns 16 rows x 128 cols;
// each wave owns one 16x16 C tile: 8x v_wmma_f32_16x16x32_f16 (K=128, 2 A mats).
__global__ __launch_bounds__(256) void sage_wmma_kernel(
    const _Float16* __restrict__ hin, const float* __restrict__ agg,
    const float* __restrict__ cnt,
    const _Float16* __restrict__ WhL, const _Float16* __restrict__ WhR,
    const float* __restrict__ bias,
    const float* __restrict__ bng, const float* __restrict__ bnb,
    const float* __restrict__ bnm, const float* __restrict__ bnv,
    _Float16* __restrict__ hout) {
  __shared__ __align__(32) _Float16 sMean[16 * HC];
  __shared__ __align__(32) _Float16 sH[16 * HC];
  __shared__ float sInv[16];

  const int rowBase = blockIdx.x * 16;
  const int tid  = threadIdx.x;
  const int lane = tid & 31;
  const int wave = tid >> 5;

  if (tid < 16) {
    float c = cnt[rowBase + tid];
    sInv[tid] = 1.0f / fmaxf(c, 1.0f);
  }
  __syncthreads();

  // Staging: 16 rows x 128 f16 each. 256 chunks of 8 f16 -> one iteration.
  //   sH:    straight 16B copy (global_load_b128 -> ds_store_b128)
  //   sMean: two float4 loads, scale by 1/cnt, pack 8 f16, one ds_store_b128
  {
    const int i = tid * 8;               // element index, tid < 256
    const int r = i >> 7;
    *(v8h*)&sH[i] = *(const v8h*)(hin + (long long)rowBase * HC + i);
    const float inv = sInv[r];
    const float4 a0 = *(const float4*)(agg + (long long)rowBase * HC + i);
    const float4 a1 = *(const float4*)(agg + (long long)rowBase * HC + i + 4);
    v8h m;
    m[0] = (_Float16)(a0.x * inv); m[1] = (_Float16)(a0.y * inv);
    m[2] = (_Float16)(a0.z * inv); m[3] = (_Float16)(a0.w * inv);
    m[4] = (_Float16)(a1.x * inv); m[5] = (_Float16)(a1.y * inv);
    m[6] = (_Float16)(a1.z * inv); m[7] = (_Float16)(a1.w * inv);
    *(v8h*)&sMean[i] = m;
  }
  __syncthreads();

  const int colBase = wave * 16;
  const int col  = lane & 15;
  const int row  = lane & 15;
  const int half = lane >> 4;
  const int gcol = colBase + col;

  // Warm this wave's weight rows in cache (global_prefetch_b8 on gfx1250).
  __builtin_prefetch(&WhL[gcol * HC], 0, 1);
  __builtin_prefetch(&WhR[gcol * HC], 0, 1);

  v8f c = {};
#pragma unroll
  for (int kt = 0; kt < 4; ++kt) {
    const int kOff = kt * 32;

    // A fragments (16-bit A 16x32): lane=row; element e -> K = kOff + 16*half + e
    // (same K order as B)
    // -> two aligned 16B chunks from LDS (ds_load_b128 each).
    const int aOff = row * HC + kOff + half * 16;
    const v8h aMlo = *(const v8h*)&sMean[aOff];
    const v8h aMhi = *(const v8h*)&sMean[aOff + 8];
    const v8h aHlo = *(const v8h*)&sH[aOff];
    const v8h aHhi = *(const v8h*)&sH[aOff + 8];
    const v16h aM = __builtin_shufflevector(aMlo, aMhi,
        0,1,2,3,4,5,6,7,8,9,10,11,12,13,14,15);
    const v16h aH = __builtin_shufflevector(aHlo, aHhi,
        0,1,2,3,4,5,6,7,8,9,10,11,12,13,14,15);

    // B fragments (B 32x16): lane=col; element e -> K = kOff + e + 16*half,
    // i.e. 16 consecutive K of channel gcol -> one aligned v16h (2x b128).
    const int bOff = gcol * HC + kOff + half * 16;
    const v16h bL = *(const v16h*)&WhL[bOff];
    const v16h bR = *(const v16h*)&WhR[bOff];

    c = __builtin_amdgcn_wmma_f32_16x16x32_f16(false, aM, false, bL,
                                               (short)0, c, false, false);
    c = __builtin_amdgcn_wmma_f32_16x16x32_f16(false, aH, false, bR,
                                               (short)0, c, false, false);
  }

  // Epilogue on the accumulator (C layout: VGPR v -> row v + 8*(lane>=16)).
  const float bl    = bias[gcol];
  const float scale = bng[gcol] * rsqrtf(bnv[gcol] + 1e-5f);
  const float mB    = bnm[gcol];
  const float bB    = bnb[gcol];
#pragma unroll
  for (int v = 0; v < 8; ++v) {
    int r = rowBase + v + (half ? 8 : 0);
    float val = c[v] + bl;
    val = (val - mB) * scale + bB;
    val = fmaxf(val, 0.0f);
    hout[r * HC + gcol] = (_Float16)val;
  }
}

// ---------------------------------------------------------------------------
// Head: out[n] = min + (max-min)*sigmoid(dot(h2[n], lin_w) + lin_b)
__global__ void head_kernel(const _Float16* __restrict__ h2,
                            const float* __restrict__ lw,
                            const float* __restrict__ lb,
                            const float* __restrict__ minr,
                            const float* __restrict__ maxr,
                            float* __restrict__ out, int N) {
  int n = blockIdx.x * blockDim.x + threadIdx.x;
  if (n >= N) return;
  float acc = lb[0];
  const v8h*    hv = (const v8h*)(h2 + (long long)n * HC);
  const float4* wv = (const float4*)lw;
#pragma unroll 4
  for (int i = 0; i < HC / 8; ++i) {
    const v8h a = hv[i];
    const float4 w0 = wv[i * 2], w1 = wv[i * 2 + 1];
    acc += (float)a[0] * w0.x + (float)a[1] * w0.y +
           (float)a[2] * w0.z + (float)a[3] * w0.w +
           (float)a[4] * w1.x + (float)a[5] * w1.y +
           (float)a[6] * w1.z + (float)a[7] * w1.w;
  }
  float sig = 1.0f / (1.0f + __expf(-acc));
  float mn = minr[0], mx = maxr[0];
  out[n] = mn + (mx - mn) * sig;
}

// ---------------------------------------------------------------------------
extern "C" void kernel_launch(void* const* d_in, const int* in_sizes, int n_in,
                              void* d_out, int out_size, void* d_ws, size_t ws_size,
                              hipStream_t stream) {
  const int N = in_sizes[0] / ING;   // 50000 (multiple of 16)
  const int E = in_sizes[20] / 2;    // 800000

  const float* x     = (const float*)d_in[0];
  const float* gemb  = (const float*)d_in[1];
  const float* W_l0  = (const float*)d_in[2];
  const float* b_l0  = (const float*)d_in[3];
  const float* W_r0  = (const float*)d_in[4];
  const float* W_l1  = (const float*)d_in[5];
  const float* b_l1  = (const float*)d_in[6];
  const float* W_r1  = (const float*)d_in[7];
  const float* bn_g0 = (const float*)d_in[8];
  const float* bn_b0 = (const float*)d_in[9];
  const float* bn_m0 = (const float*)d_in[10];
  const float* bn_v0 = (const float*)d_in[11];
  const float* bn_g1 = (const float*)d_in[12];
  const float* bn_b1 = (const float*)d_in[13];
  const float* bn_m1 = (const float*)d_in[14];
  const float* bn_v1 = (const float*)d_in[15];
  const float* lin_w = (const float*)d_in[16];
  const float* lin_b = (const float*)d_in[17];
  const float* minr  = (const float*)d_in[18];
  const float* maxr  = (const float*)d_in[19];
  const int*   ei    = (const int*)d_in[20];
  const int*   gids  = (const int*)d_in[21];
  const unsigned char* gmask = (const unsigned char*)d_in[22];

  // Workspace layout (256B aligned blocks)
  char* ws = (char*)d_ws;
  size_t off = 0;
  auto alloc = [&](size_t bytes) {
    void* p = ws + off;
    off += (bytes + 255) & ~(size_t)255;
    return p;
  };
  _Float16* WhL0 = (_Float16*)alloc(HC * HC * sizeof(_Float16));
  _Float16* WhR0 = (_Float16*)alloc(HC * HC * sizeof(_Float16));
  _Float16* WhL1 = (_Float16*)alloc(HC * HC * sizeof(_Float16));
  _Float16* WhR1 = (_Float16*)alloc(HC * HC * sizeof(_Float16));
  _Float16* h0 = (_Float16*)alloc((size_t)N * HC * sizeof(_Float16));
  _Float16* h1 = (_Float16*)alloc((size_t)N * HC * sizeof(_Float16));
  _Float16* h2 = (_Float16*)alloc((size_t)N * HC * sizeof(_Float16));
  float* agg = (float*)alloc((size_t)N * HC * sizeof(float));
  float* cnt = (float*)alloc((size_t)N * sizeof(float));
  (void)ws_size; (void)n_in; (void)out_size;

  const int T = 256;
  // Weight f16 conversion (row-major preserved)
  wprep_kernel<<<(HC * HC / 4 + T - 1) / T, T, 0, stream>>>(W_l0, WhL0);
  wprep_kernel<<<(HC * HC / 4 + T - 1) / T, T, 0, stream>>>(W_r0, WhR0);
  wprep_kernel<<<(HC * HC / 4 + T - 1) / T, T, 0, stream>>>(W_l1, WhL1);
  wprep_kernel<<<(HC * HC / 4 + T - 1) / T, T, 0, stream>>>(W_r1, WhR1);

  // Feature assembly
  pool_concat_kernel<<<((long long)N * HC + T - 1) / T, T, 0, stream>>>(
      x, gemb, gids, gmask, h0, N);

  const long long aggN = (long long)N * HC;
  const long long scatThreads = (long long)E * 16;

  // ---- Layer 0 ----
  zero_kernel<<<(aggN + T - 1) / T, T, 0, stream>>>(agg, aggN);
  zero_kernel<<<(N + T - 1) / T, T, 0, stream>>>(cnt, N);
  edge_scatter_kernel<<<(scatThreads + T - 1) / T, T, 0, stream>>>(h0, ei, agg, cnt, E);
  sage_wmma_kernel<<<N / 16, 256, 0, stream>>>(h0, agg, cnt, WhL0, WhR0, b_l0,
                                               bn_g0, bn_b0, bn_m0, bn_v0, h1);

  // ---- Layer 1 ----
  zero_kernel<<<(aggN + T - 1) / T, T, 0, stream>>>(agg, aggN);
  zero_kernel<<<(N + T - 1) / T, T, 0, stream>>>(cnt, N);
  edge_scatter_kernel<<<(scatThreads + T - 1) / T, T, 0, stream>>>(h1, ei, agg, cnt, E);
  sage_wmma_kernel<<<N / 16, 256, 0, stream>>>(h1, agg, cnt, WhL1, WhR1, b_l1,
                                               bn_g1, bn_b1, bn_m1, bn_v1, h2);

  // ---- Head ----
  head_kernel<<<(N + T - 1) / T, T, 0, stream>>>(h2, lin_w, lin_b, minr, maxr,
                                                 (float*)d_out, N);
}
